// MOLEModule_10222022164562
// MI455X (gfx1250) — compile-verified, matrix operand order checked by reading, not run on hardware
//
#include <hip/hip_runtime.h>
#include <hip/hip_bf16.h>

// ---------------- problem constants ----------------
#define N_TOK 4096
#define D_DIM 1024
#define F_DIM 4096
#define E_EXP 32
#define K_TOP 2
#define CAP   512

// ---------------- GEMM tiling ----------------
#define BM 128
#define BN 128
#define BK 32
#define KP 40          // padded K stride in LDS (bf16 elems) -> conflict-free b128 reads

typedef __attribute__((ext_vector_type(16))) __bf16 v16bf;
typedef __attribute__((ext_vector_type(8)))  __bf16 v8bf;
typedef __attribute__((ext_vector_type(4)))  __bf16 v4bf;
typedef __attribute__((ext_vector_type(8)))  float  v8f;
typedef __attribute__((ext_vector_type(4)))  int    v4i;

// ---- gfx1250 async global->LDS copy (probe-guarded; falls back to reg copy) ----
#if defined(__gfx1250__) && __has_builtin(__builtin_amdgcn_global_load_async_to_lds_b128)
#define HAS_ASYNC_LDS 1
#else
#define HAS_ASYNC_LDS 0
#endif

#if HAS_ASYNC_LDS
typedef __attribute__((address_space(1))) v4i as1_v4i;   // global
typedef __attribute__((address_space(3))) v4i as3_v4i;   // LDS
__device__ __forceinline__ void async_cp16(const __bf16* g, __bf16* l) {
    __builtin_amdgcn_global_load_async_to_lds_b128(
        (as1_v4i*)(g), (as3_v4i*)(l), 0, 0);
}
#if __has_builtin(__builtin_amdgcn_s_wait_asynccnt)
__device__ __forceinline__ void wait_async() { __builtin_amdgcn_s_wait_asynccnt(0); }
#else
__device__ __forceinline__ void wait_async() { asm volatile("s_wait_asynccnt 0x0" ::: "memory"); }
#endif
#else
__device__ __forceinline__ void wait_async() {}
#endif

// ---------------- helpers ----------------
__device__ __forceinline__ float gelu_tanh(float x) {
    const float c0 = 0.7978845608028654f;   // sqrt(2/pi)
    const float c1 = 0.044715f;
    float u = c0 * (x + c1 * x * x * x);
    return 0.5f * x * (1.0f + tanhf(u));
}

// Load one 16x32 bf16 fragment from an LDS tile stored row-major with KP
// stride. Per ISA layout: lanes 0-15 hold K=0..7,16..23; lanes 16-31 hold
// K=8..15,24..31.
__device__ __forceinline__ v16bf frag_ld(const __bf16* tile, int lane) {
    int r  = lane & 15;
    int ko = (lane & 16) ? 8 : 0;
    const __bf16* p = tile + r * KP + ko;
    v8bf lo = *(const v8bf*)(p);
    v8bf hi = *(const v8bf*)(p + 16);
    return __builtin_shufflevector(lo, hi, 0,1,2,3,4,5,6,7,8,9,10,11,12,13,14,15);
}

// ============ 1. gating: logits -> softmax -> top2 -> slot assign ============
// one wave32 per token; lane == expert (E==32 exactly fills a wave)
__global__ __launch_bounds__(256) void gate_kernel(
    const float* __restrict__ x, const float* __restrict__ gw,
    const float* __restrict__ gb, int* __restrict__ counters,
    int* __restrict__ tok_of, float* __restrict__ wgt_of) {
    int wave = threadIdx.x >> 5;
    int lane = threadIdx.x & 31;
    int n = blockIdx.x * 8 + wave;

    const float* xr = x  + (size_t)n    * D_DIM;
    const float* wr = gw + (size_t)lane * D_DIM;
    float acc = 0.0f;
    for (int d = 0; d < D_DIM; d += 4) {
        float4 xv = *(const float4*)(xr + d);   // same addr all lanes: broadcast
        float4 wv = *(const float4*)(wr + d);
        acc += xv.x * wv.x + xv.y * wv.y + xv.z * wv.z + xv.w * wv.w;
    }
    float logit = acc + gb[lane];

    // softmax across the wave
    float m = logit;
    for (int o = 16; o; o >>= 1) m = fmaxf(m, __shfl_xor(m, o, 32));
    float p = __expf(logit - m);
    float s = p;
    for (int o = 16; o; o >>= 1) s += __shfl_xor(s, o, 32);
    float prob = p / s;

    // top-1 (lowest index wins ties, matching lax.top_k)
    float p1 = prob; int i1 = lane;
    for (int o = 16; o; o >>= 1) {
        float po = __shfl_xor(p1, o, 32); int io = __shfl_xor(i1, o, 32);
        if (po > p1 || (po == p1 && io < i1)) { p1 = po; i1 = io; }
    }
    // top-2
    float p2 = (lane == i1) ? -1.0f : prob; int i2 = lane;
    for (int o = 16; o; o >>= 1) {
        float po = __shfl_xor(p2, o, 32); int io = __shfl_xor(i2, o, 32);
        if (po > p2 || (po == p2 && io < i2)) { p2 = po; i2 = io; }
    }
    float inv = 1.0f / (p1 + p2);
    if (lane == 0) {
        int pos = atomicAdd(&counters[i1], 1);
        if (pos < CAP) { tok_of[i1 * CAP + pos] = n; wgt_of[i1 * CAP + pos] = p1 * inv; }
        pos = atomicAdd(&counters[i2], 1);
        if (pos < CAP) { tok_of[i2 * CAP + pos] = n; wgt_of[i2 * CAP + pos] = p2 * inv; }
    }
}

// ============ 2. dispatch: gather token rows into bf16 [E,C,D] ============
__global__ __launch_bounds__(128) void dispatch_kernel(
    const float* __restrict__ x, const int* __restrict__ tok_of,
    __bf16* __restrict__ xin) {
    int slot = blockIdx.x;            // e*CAP + c
    int tok  = tok_of[slot];
    __bf16* dst = xin + (size_t)slot * D_DIM;
    int base = threadIdx.x * 8;       // 128 thr * 8 = 1024 = D
    v8bf v;
    if (tok < 0) {
        #pragma unroll
        for (int i = 0; i < 8; ++i) v[i] = (__bf16)0.0f;
    } else {
        const float* src = x + (size_t)tok * D_DIM + base;
        float4 a = *(const float4*)(src);
        float4 b = *(const float4*)(src + 4);
        v[0]=(__bf16)a.x; v[1]=(__bf16)a.y; v[2]=(__bf16)a.z; v[3]=(__bf16)a.w;
        v[4]=(__bf16)b.x; v[5]=(__bf16)b.y; v[6]=(__bf16)b.z; v[7]=(__bf16)b.w;
    }
    *(v8bf*)(dst + base) = v;
}

// ============ 3. GEMM1: H = gelu(Xe[C,D] @ w1[e]^T + b1) ============
// Double-buffered LDS pipeline; A tile copied via async-to-LDS (if available),
// B tile (fp32 weights) register-staged and converted to bf16 in flight.
__global__ __launch_bounds__(256) void ffn1_kernel(
    const __bf16* __restrict__ xin, const float* __restrict__ w1,
    const float* __restrict__ b1, __bf16* __restrict__ H) {
    __shared__ __bf16 lds_a[2][BM * KP];
    __shared__ __bf16 lds_b[2][BN * KP];
    const int e   = blockIdx.z;
    const int n0  = blockIdx.x * BN;
    const int m0  = blockIdx.y * BM;
    const int tid = threadIdx.x;
    const int lane = tid & 31, wave = tid >> 5;
    const int wm = (wave >> 2) * 64;     // 2 waves in M
    const int wn = (wave & 3) * 32;      // 4 waves in N
    const float* w1e = w1 + (size_t)e * F_DIM * D_DIM;
    const __bf16* Ab = xin + (size_t)(e * CAP + m0) * D_DIM;

    // per-thread static chunk coords
    const int ar0 = (tid)       >> 2, ako0 = ((tid)       & 3) * 8;
    const int ar1 = (tid + 256) >> 2, ako1 = ((tid + 256) & 3) * 8;
    int br[4], bko[4];
    #pragma unroll
    for (int i = 0; i < 4; ++i) { int c = tid + i * 256; br[i] = c >> 3; bko[i] = (c & 7) * 4; }

    v8f acc[4][2];
    #pragma unroll
    for (int i = 0; i < 4; ++i)
        #pragma unroll
        for (int j = 0; j < 2; ++j) acc[i][j] = (v8f)0.0f;

    float4 breg[4];
#if !HAS_ASYNC_LDS
    v8bf areg[2];
#endif
    // ---- prologue: stage k0 = 0 into buffer 0 ----
#if HAS_ASYNC_LDS
    async_cp16(Ab + (size_t)ar0 * D_DIM + ako0, &lds_a[0][ar0 * KP + ako0]);
    async_cp16(Ab + (size_t)ar1 * D_DIM + ako1, &lds_a[0][ar1 * KP + ako1]);
#else
    *(v8bf*)(&lds_a[0][ar0 * KP + ako0]) = *(const v8bf*)(Ab + (size_t)ar0 * D_DIM + ako0);
    *(v8bf*)(&lds_a[0][ar1 * KP + ako1]) = *(const v8bf*)(Ab + (size_t)ar1 * D_DIM + ako1);
#endif
    #pragma unroll
    for (int i = 0; i < 4; ++i)
        breg[i] = *(const float4*)(w1e + (size_t)(n0 + br[i]) * D_DIM + bko[i]);
    #pragma unroll
    for (int i = 0; i < 4; ++i) {
        v4bf b = { (__bf16)breg[i].x, (__bf16)breg[i].y, (__bf16)breg[i].z, (__bf16)breg[i].w };
        *(v4bf*)(&lds_b[0][br[i] * KP + bko[i]]) = b;
    }
    wait_async();
    __syncthreads();

    int buf = 0;
    for (int k0 = 0; k0 < D_DIM; k0 += BK, buf ^= 1) {
        const bool nx = (k0 + BK) < D_DIM;
        if (nx) {
            const int kn = k0 + BK;
#if HAS_ASYNC_LDS
            async_cp16(Ab + (size_t)ar0 * D_DIM + kn + ako0, &lds_a[buf ^ 1][ar0 * KP + ako0]);
            async_cp16(Ab + (size_t)ar1 * D_DIM + kn + ako1, &lds_a[buf ^ 1][ar1 * KP + ako1]);
#else
            areg[0] = *(const v8bf*)(Ab + (size_t)ar0 * D_DIM + kn + ako0);
            areg[1] = *(const v8bf*)(Ab + (size_t)ar1 * D_DIM + kn + ako1);
#endif
            #pragma unroll
            for (int i = 0; i < 4; ++i) {
                const float* src = w1e + (size_t)(n0 + br[i]) * D_DIM + kn + bko[i];
                breg[i] = *(const float4*)(src);
                if (kn + BK < D_DIM) __builtin_prefetch(src + BK, 0, 1);  // L2 warm-up
            }
        }
        v16bf af[4], bf_[2];
        #pragma unroll
        for (int mt = 0; mt < 4; ++mt) af[mt]  = frag_ld(&lds_a[buf][(wm + mt * 16) * KP], lane);
        #pragma unroll
        for (int nt = 0; nt < 2; ++nt) bf_[nt] = frag_ld(&lds_b[buf][(wn + nt * 16) * KP], lane);
        #pragma unroll
        for (int mt = 0; mt < 4; ++mt)
            #pragma unroll
            for (int nt = 0; nt < 2; ++nt)
                acc[mt][nt] = __builtin_amdgcn_wmma_f32_16x16x32_bf16(
                    false, af[mt], false, bf_[nt], (short)0, acc[mt][nt], false, false);
        if (nx) {
#if !HAS_ASYNC_LDS
            *(v8bf*)(&lds_a[buf ^ 1][ar0 * KP + ako0]) = areg[0];
            *(v8bf*)(&lds_a[buf ^ 1][ar1 * KP + ako1]) = areg[1];
#endif
            #pragma unroll
            for (int i = 0; i < 4; ++i) {
                v4bf b = { (__bf16)breg[i].x, (__bf16)breg[i].y,
                           (__bf16)breg[i].z, (__bf16)breg[i].w };
                *(v4bf*)(&lds_b[buf ^ 1][br[i] * KP + bko[i]]) = b;
            }
            wait_async();
        }
        __syncthreads();
    }
    // epilogue: bias + gelu -> bf16 H
    const int nl = lane & 15;
    const int rb = (lane & 16) ? 8 : 0;
    #pragma unroll
    for (int mt = 0; mt < 4; ++mt)
        #pragma unroll
        for (int nt = 0; nt < 2; ++nt) {
            int gn = n0 + wn + nt * 16 + nl;
            float bias = b1[e * F_DIM + gn];
            #pragma unroll
            for (int v = 0; v < 8; ++v) {
                int gm = m0 + wm + mt * 16 + rb + v;
                float h = gelu_tanh(acc[mt][nt][v] + bias);
                H[((size_t)(e * CAP + gm)) * F_DIM + gn] = (__bf16)h;
            }
        }
}

// ============ 4. GEMM2 + combine: out[tok] += w * (H[C,F] @ w2[e][F,D] + b2) ====
// Same pipeline; w2 is K-major so the bf16 convert also transposes into LDS.
__global__ __launch_bounds__(256) void ffn2_kernel(
    const __bf16* __restrict__ H, const float* __restrict__ w2,
    const float* __restrict__ b2, const int* __restrict__ tok_of,
    const float* __restrict__ wgt_of, float* __restrict__ out) {
    __shared__ __bf16 lds_a[2][BM * KP];
    __shared__ __bf16 lds_b[2][BN * KP];
    const int e   = blockIdx.z;
    const int n0  = blockIdx.x * BN;
    const int m0  = blockIdx.y * BM;
    const int tid = threadIdx.x;
    const int lane = tid & 31, wave = tid >> 5;
    const int wm = (wave >> 2) * 64;
    const int wn = (wave & 3) * 32;
    const float* w2e = w2 + (size_t)e * F_DIM * D_DIM;
    const __bf16* Ab = H + (size_t)(e * CAP + m0) * F_DIM;

    const int ar0 = (tid)       >> 2, ako0 = ((tid)       & 3) * 8;
    const int ar1 = (tid + 256) >> 2, ako1 = ((tid + 256) & 3) * 8;
    int bkr[4], bnc[4];
    #pragma unroll
    for (int i = 0; i < 4; ++i) { int c = tid + i * 256; bkr[i] = c >> 5; bnc[i] = (c & 31) * 4; }

    v8f acc[4][2];
    #pragma unroll
    for (int i = 0; i < 4; ++i)
        #pragma unroll
        for (int j = 0; j < 2; ++j) acc[i][j] = (v8f)0.0f;

    float4 breg[4];
#if !HAS_ASYNC_LDS
    v8bf areg[2];
#endif
    // ---- prologue: stage k0 = 0 into buffer 0 ----
#if HAS_ASYNC_LDS
    async_cp16(Ab + (size_t)ar0 * F_DIM + ako0, &lds_a[0][ar0 * KP + ako0]);
    async_cp16(Ab + (size_t)ar1 * F_DIM + ako1, &lds_a[0][ar1 * KP + ako1]);
#else
    *(v8bf*)(&lds_a[0][ar0 * KP + ako0]) = *(const v8bf*)(Ab + (size_t)ar0 * F_DIM + ako0);
    *(v8bf*)(&lds_a[0][ar1 * KP + ako1]) = *(const v8bf*)(Ab + (size_t)ar1 * F_DIM + ako1);
#endif
    #pragma unroll
    for (int i = 0; i < 4; ++i)
        breg[i] = *(const float4*)(w2e + (size_t)bkr[i] * D_DIM + n0 + bnc[i]);
    #pragma unroll
    for (int i = 0; i < 4; ++i) {
        lds_b[0][(bnc[i] + 0) * KP + bkr[i]] = (__bf16)breg[i].x;
        lds_b[0][(bnc[i] + 1) * KP + bkr[i]] = (__bf16)breg[i].y;
        lds_b[0][(bnc[i] + 2) * KP + bkr[i]] = (__bf16)breg[i].z;
        lds_b[0][(bnc[i] + 3) * KP + bkr[i]] = (__bf16)breg[i].w;
    }
    wait_async();
    __syncthreads();

    int buf = 0;
    for (int k0 = 0; k0 < F_DIM; k0 += BK, buf ^= 1) {
        const bool nx = (k0 + BK) < F_DIM;
        if (nx) {
            const int kn = k0 + BK;
#if HAS_ASYNC_LDS
            async_cp16(Ab + (size_t)ar0 * F_DIM + kn + ako0, &lds_a[buf ^ 1][ar0 * KP + ako0]);
            async_cp16(Ab + (size_t)ar1 * F_DIM + kn + ako1, &lds_a[buf ^ 1][ar1 * KP + ako1]);
#else
            areg[0] = *(const v8bf*)(Ab + (size_t)ar0 * F_DIM + kn + ako0);
            areg[1] = *(const v8bf*)(Ab + (size_t)ar1 * F_DIM + kn + ako1);
#endif
            #pragma unroll
            for (int i = 0; i < 4; ++i) {
                const float* src = w2e + (size_t)(kn + bkr[i]) * D_DIM + n0 + bnc[i];
                breg[i] = *(const float4*)(src);
                if (kn + BK < F_DIM)
                    __builtin_prefetch(src + (size_t)BK * D_DIM, 0, 1);  // L2 warm-up
            }
        }
        v16bf af[4], bf_[2];
        #pragma unroll
        for (int mt = 0; mt < 4; ++mt) af[mt]  = frag_ld(&lds_a[buf][(wm + mt * 16) * KP], lane);
        #pragma unroll
        for (int nt = 0; nt < 2; ++nt) bf_[nt] = frag_ld(&lds_b[buf][(wn + nt * 16) * KP], lane);
        #pragma unroll
        for (int mt = 0; mt < 4; ++mt)
            #pragma unroll
            for (int nt = 0; nt < 2; ++nt)
                acc[mt][nt] = __builtin_amdgcn_wmma_f32_16x16x32_bf16(
                    false, af[mt], false, bf_[nt], (short)0, acc[mt][nt], false, false);
        if (nx) {
#if !HAS_ASYNC_LDS
            *(v8bf*)(&lds_a[buf ^ 1][ar0 * KP + ako0]) = areg[0];
            *(v8bf*)(&lds_a[buf ^ 1][ar1 * KP + ako1]) = areg[1];
#endif
            #pragma unroll
            for (int i = 0; i < 4; ++i) {
                lds_b[buf ^ 1][(bnc[i] + 0) * KP + bkr[i]] = (__bf16)breg[i].x;
                lds_b[buf ^ 1][(bnc[i] + 1) * KP + bkr[i]] = (__bf16)breg[i].y;
                lds_b[buf ^ 1][(bnc[i] + 2) * KP + bkr[i]] = (__bf16)breg[i].z;
                lds_b[buf ^ 1][(bnc[i] + 3) * KP + bkr[i]] = (__bf16)breg[i].w;
            }
            wait_async();
        }
        __syncthreads();
    }
    // fused combine: atomicAdd into out[token] (K=2 experts merge via atomics)
    const int nl = lane & 15;
    const int rb = (lane & 16) ? 8 : 0;
    #pragma unroll
    for (int mt = 0; mt < 4; ++mt)
        #pragma unroll
        for (int v = 0; v < 8; ++v) {
            int gm = m0 + wm + mt * 16 + rb + v;
            int tok = tok_of[e * CAP + gm];
            if (tok < 0) continue;
            float wgt = wgt_of[e * CAP + gm];
            #pragma unroll
            for (int nt = 0; nt < 2; ++nt) {
                int gn = n0 + wn + nt * 16 + nl;
                float val = acc[mt][nt][v] + b2[e * D_DIM + gn];
                atomicAdd(out + (size_t)tok * D_DIM + gn, wgt * val);
            }
        }
}

// ---------------- launch ----------------
extern "C" void kernel_launch(void* const* d_in, const int* in_sizes, int n_in,
                              void* d_out, int out_size, void* d_ws, size_t ws_size,
                              hipStream_t stream) {
    (void)in_sizes; (void)n_in; (void)out_size; (void)ws_size;
    const float* x  = (const float*)d_in[0];
    const float* gw = (const float*)d_in[1];
    const float* gb = (const float*)d_in[2];
    const float* w1 = (const float*)d_in[3];
    const float* b1 = (const float*)d_in[4];
    const float* w2 = (const float*)d_in[5];
    const float* b2 = (const float*)d_in[6];
    float* out = (float*)d_out;

    char* ws = (char*)d_ws;
    size_t off = 0;
    int*    counters = (int*)(ws + off);               off += 256;
    int*    tok_of   = (int*)(ws + off);               off += (size_t)E_EXP * CAP * sizeof(int);
    float*  wgt_of   = (float*)(ws + off);             off += (size_t)E_EXP * CAP * sizeof(float);
    __bf16* xin      = (__bf16*)(ws + off);            off += (size_t)E_EXP * CAP * D_DIM * 2;
    __bf16* Hbuf     = (__bf16*)(ws + off);            // E*CAP*F_DIM*2 = 128 MB

    (void)hipMemsetAsync(out, 0, (size_t)N_TOK * D_DIM * sizeof(float), stream);
    (void)hipMemsetAsync(counters, 0, E_EXP * sizeof(int), stream);
    (void)hipMemsetAsync(tok_of, 0xFF, (size_t)E_EXP * CAP * sizeof(int), stream);  // -1

    gate_kernel<<<N_TOK / 8, 256, 0, stream>>>(x, gw, gb, counters, tok_of, wgt_of);
    dispatch_kernel<<<E_EXP * CAP, 128, 0, stream>>>(x, tok_of, xin);
    ffn1_kernel<<<dim3(F_DIM / BN, CAP / BM, E_EXP), 256, 0, stream>>>(xin, w1, b1, Hbuf);
    ffn2_kernel<<<dim3(D_DIM / BN, CAP / BM, E_EXP), 256, 0, stream>>>(Hbuf, w2, b2,
                                                                       tok_of, wgt_of, out);
}
